// GatModule_39951785787637
// MI455X (gfx1250) — matrix operand order repaired and checked by reading, not compiled, not run on hardware
//
#include <hip/hip_runtime.h>
#include <hip/hip_bf16.h>

#define NN   10000
#define MP   10016          // M padded to multiple of 32 (313 row tiles of 32)
#define RT32 313
#define EE   100000
#define EDGE 110000
#define CC   200

typedef __attribute__((ext_vector_type(16))) __bf16 v16bf;
typedef __attribute__((ext_vector_type(8)))  float  v8f;

// ---------- helpers ----------

__device__ __forceinline__ unsigned enc_f(float f) {
    unsigned u = __float_as_uint(f);
    return (u & 0x80000000u) ? ~u : (u | 0x80000000u);
}
__device__ __forceinline__ float dec_f(unsigned k) {
    unsigned u = (k & 0x80000000u) ? (k & 0x7FFFFFFFu) : ~k;
    return __uint_as_float(u);
}
__device__ __forceinline__ float gelu_exact(float v) {
    return 0.5f * v * (1.0f + erff(v * 0.70710678118654752f));
}
__device__ __forceinline__ void edge_sd(int e, const int* ei, int& s, int& d) {
    if (e < EE) { s = ei[e]; d = ei[EE + e]; }
    else        { s = e - EE; d = e - EE; }
}

// ---------- utility kernels ----------

__global__ void k_fill_u32(unsigned* p, unsigned v, int n) {
    int i = blockIdx.x * blockDim.x + threadIdx.x;
    if (i < n) p[i] = v;
}

// X (N x Fin, f32) -> Xb (N x Kp, bf16), K-pad zeroed (pad rows of Xb are never read validly)
__global__ void k_x_to_bf16(const float* X, __bf16* Xb, int Fin, int Kp) {
    int i = blockIdx.x * blockDim.x + threadIdx.x;
    if (i >= NN * Kp) return;
    int n = i / Kp, j = i % Kp;
    Xb[i] = (j < Fin) ? (__bf16)X[n * Fin + j] : (__bf16)0.0f;
}

// W (Fin x Fout, f32, row-major) -> Wt (FoutP x Kp, bf16) = transpose, zero-padded
__global__ void k_wt_bf16(const float* W, __bf16* Wt, int Fin, int Fout, int Kp, int FoutP) {
    int i = blockIdx.x * blockDim.x + threadIdx.x;
    if (i >= FoutP * Kp) return;
    int o = i / Kp, k = i % Kp;
    Wt[i] = (o < Fout && k < Fin) ? (__bf16)W[k * Fout + o] : (__bf16)0.0f;
}

// ---------- WMMA GEMM: H = Xb (MP x Kp) * Wt^T (Kp x FoutP) ----------
// One wave -> 32x64 tile: 2 A-frags x 4 B-frags -> 8 WMMAs per K=32 step
// from 12 b128 loads (1.5 loads/WMMA). 8 waves/block, each an independent tile.
__global__ void __launch_bounds__(256)
k_gemm_wmma(const __bf16* __restrict__ Xb, const __bf16* __restrict__ Wt,
            float* __restrict__ Hout, int Kp, int FoutP) {
    const int lane = threadIdx.x & 31;
    const int wave = threadIdx.x >> 5;
    const int rt = blockIdx.y * 8 + wave;               // 313 row tiles of 32
    if (rt >= RT32) return;                             // uniform per wave
    const int colBase = blockIdx.x * 64;

    const int hi  = lane >> 4;                          // lane half
    const int l15 = lane & 15;
    const int aK0 = hi << 3;                            // A: K 0-7 / 8-15 (+16 second chunk)
    const int bK0 = hi << 3;                            // B: K 0-7 / 8-15 (+16 second chunk)

    const __bf16* aP0 = Xb + (size_t)(rt * 32 + l15) * Kp + aK0;
    const __bf16* aP1 = aP0 + (size_t)16 * Kp;
    const __bf16* bP0 = Wt + (size_t)(colBase + l15) * Kp + bK0;
    const __bf16* bP1 = bP0 + (size_t)16 * Kp;
    const __bf16* bP2 = bP0 + (size_t)32 * Kp;
    const __bf16* bP3 = bP0 + (size_t)48 * Kp;

    v8f acc[8] = {};                                    // [rowfrag*4 + colfrag]

    for (int k = 0; k < Kp; k += 32) {
        union U { v16bf v; float4 f4[2]; } a0, a1, b0, b1, b2, b3;
        a0.f4[0] = *(const float4*)(aP0 + k);
        a0.f4[1] = *(const float4*)(aP0 + k + 16);
        a1.f4[0] = *(const float4*)(aP1 + k);
        a1.f4[1] = *(const float4*)(aP1 + k + 16);
        b0.f4[0] = *(const float4*)(bP0 + k);
        b0.f4[1] = *(const float4*)(bP0 + k + 16);
        b1.f4[0] = *(const float4*)(bP1 + k);
        b1.f4[1] = *(const float4*)(bP1 + k + 16);
        b2.f4[0] = *(const float4*)(bP2 + k);
        b2.f4[1] = *(const float4*)(bP2 + k + 16);
        b3.f4[0] = *(const float4*)(bP3 + k);
        b3.f4[1] = *(const float4*)(bP3 + k + 16);
        acc[0] = __builtin_amdgcn_wmma_f32_16x16x32_bf16(false, a0.v, false, b0.v, (short)0, acc[0], false, false);
        acc[1] = __builtin_amdgcn_wmma_f32_16x16x32_bf16(false, a0.v, false, b1.v, (short)0, acc[1], false, false);
        acc[2] = __builtin_amdgcn_wmma_f32_16x16x32_bf16(false, a0.v, false, b2.v, (short)0, acc[2], false, false);
        acc[3] = __builtin_amdgcn_wmma_f32_16x16x32_bf16(false, a0.v, false, b3.v, (short)0, acc[3], false, false);
        acc[4] = __builtin_amdgcn_wmma_f32_16x16x32_bf16(false, a1.v, false, b0.v, (short)0, acc[4], false, false);
        acc[5] = __builtin_amdgcn_wmma_f32_16x16x32_bf16(false, a1.v, false, b1.v, (short)0, acc[5], false, false);
        acc[6] = __builtin_amdgcn_wmma_f32_16x16x32_bf16(false, a1.v, false, b2.v, (short)0, acc[6], false, false);
        acc[7] = __builtin_amdgcn_wmma_f32_16x16x32_bf16(false, a1.v, false, b3.v, (short)0, acc[7], false, false);
    }

    // D layout: VGPR r = row (hi?8:0)+r within 16-row frag; lane&15 = column within 16-col frag
#pragma unroll
    for (int i = 0; i < 2; ++i) {
        const int rBase = rt * 32 + i * 16 + (hi << 3);
#pragma unroll
        for (int j = 0; j < 4; ++j) {
            const int c = colBase + j * 16 + l15;
            float* out = Hout + (size_t)rBase * FoutP + c;
#pragma unroll
            for (int r = 0; r < 8; ++r)
                out[(size_t)r * FoutP] = acc[i * 4 + j][r];
        }
    }
}

// ---------- attention ----------

// alpha_src/dst[n,h] = dot(h[n,h,:], a_src/dst[h,:])
__global__ void k_alpha(const float* __restrict__ Hc, const float* __restrict__ a_src,
                        const float* __restrict__ a_dst, float* As, float* Ad,
                        int heads, int FoutP) {
    int i = blockIdx.x * blockDim.x + threadIdx.x;
    if (i >= NN * heads) return;
    int n = i / heads, hd = i % heads;
    const float* hp = Hc + (size_t)n * FoutP + hd * CC;
    const float* ps = a_src + hd * CC;
    const float* pd = a_dst + hd * CC;
    float s = 0.f, d = 0.f;
    for (int c = 0; c < CC; ++c) { float v = hp[c]; s += v * ps[c]; d += v * pd[c]; }
    As[i] = s; Ad[i] = d;
}

__global__ void k_edge_max(const int* __restrict__ ei, const float* __restrict__ As,
                           const float* __restrict__ Ad, float* Ebuf, unsigned* Mk,
                           int heads) {
    int i = blockIdx.x * blockDim.x + threadIdx.x;
    if (i >= EDGE * heads) return;
    int e = i / heads, hd = i % heads;
    int s, d; edge_sd(e, ei, s, d);
    float v = As[s * heads + hd] + Ad[d * heads + hd];
    float e_ = (v >= 0.f) ? v : 0.2f * v;               // leaky_relu 0.2
    Ebuf[i] = e_;
    atomicMax(&Mk[d * heads + hd], enc_f(e_));
}

__global__ void k_edge_exp(const int* __restrict__ ei, float* Ebuf,
                           const unsigned* __restrict__ Mk, float* Den, int heads) {
    int i = blockIdx.x * blockDim.x + threadIdx.x;
    if (i >= EDGE * heads) return;
    int e = i / heads, hd = i % heads;
    int s, d; edge_sd(e, ei, s, d); (void)s;
    float ex = expf(Ebuf[i] - dec_f(Mk[d * heads + hd]));
    Ebuf[i] = ex;
    atomicAdd(&Den[d * heads + hd], ex);
}

__global__ void k_edge_norm(const int* __restrict__ ei, const float* __restrict__ Ebuf,
                            const float* __restrict__ Den, float* Abuf, int heads,
                            float* attn_out) {
    int i = blockIdx.x * blockDim.x + threadIdx.x;
    if (i >= EDGE * heads) return;
    int e = i / heads, hd = i % heads;
    int s, d; edge_sd(e, ei, s, d); (void)s;
    float a = Ebuf[i] / Den[d * heads + hd];
    Abuf[i] = a;
    if (attn_out && heads == 1) attn_out[e] = a;        // last layer edge attention
}

// out[dst, h, c] += h[src, h, c] * alpha[e, h]  (4 channels per thread)
__global__ void k_aggregate(const int* __restrict__ ei, const float* __restrict__ Hc,
                            const float* __restrict__ Abuf, float* __restrict__ Out,
                            int heads, int FoutP) {
    int i = blockIdx.x * blockDim.x + threadIdx.x;
    const int C4 = CC / 4;
    if (i >= EDGE * heads * C4) return;
    int c4 = i % C4;
    int hd = (i / C4) % heads;
    int e  = i / (C4 * heads);
    int s, d; edge_sd(e, ei, s, d);
    float a = Abuf[e * heads + hd];
    const float* hs = Hc + (size_t)s * FoutP + hd * CC + c4 * 4;
    float* od       = Out + (size_t)d * FoutP + hd * CC + c4 * 4;
    atomicAdd(od + 0, hs[0] * a);
    atomicAdd(od + 1, hs[1] * a);
    atomicAdd(od + 2, hs[2] * a);
    atomicAdd(od + 3, hs[3] * a);
}

// bias + exact gelu; optionally emit bf16 activations for next layer and/or f32 out
__global__ void k_bias_gelu(const float* __restrict__ Out, const float* __restrict__ bias,
                            __bf16* xb_next, float* xf_out, int heads, int FoutP) {
    int i = blockIdx.x * blockDim.x + threadIdx.x;
    const int Fout = heads * CC;
    if (i >= NN * Fout) return;
    int n = i / Fout, f = i % Fout;
    float g = gelu_exact(Out[(size_t)n * FoutP + f] + bias[f]);
    if (xb_next) xb_next[(size_t)n * Fout + f] = (__bf16)g;   // stride Fout == next Kp
    if (xf_out)  xf_out[(size_t)n * Fout + f] = g;
}

// ---------- host orchestration ----------

extern "C" void kernel_launch(void* const* d_in, const int* in_sizes, int n_in,
                              void* d_out, int out_size, void* d_ws, size_t ws_size,
                              hipStream_t stream) {
    (void)in_sizes; (void)n_in; (void)out_size; (void)ws_size;
    const float* X  = (const float*)d_in[0];
    const int*   EI = (const int*)d_in[1];
    const float *W[5], *AS[5], *AD[5], *B[5];
    for (int l = 0; l < 5; ++l) {
        W[l]  = (const float*)d_in[2 + 4 * l + 0];
        AS[l] = (const float*)d_in[2 + 4 * l + 1];
        AD[l] = (const float*)d_in[2 + 4 * l + 2];
        B[l]  = (const float*)d_in[2 + 4 * l + 3];
    }

    // workspace carve (256B aligned)
    char* w = (char*)d_ws;
    auto carve = [&](size_t bytes) { char* p = w; w += (bytes + 255) & ~(size_t)255; return p; };
    __bf16*   Xb   = (__bf16*)  carve((size_t)MP * 800 * 2);   // activations, padded rows
    __bf16*   Wt   = (__bf16*)  carve((size_t)832 * 800 * 2);  // transposed weights
    float*    Hbuf = (float*)   carve((size_t)MP * 832 * 4);   // h = x@W, padded
    float*    Obuf = (float*)   carve((size_t)NN * 832 * 4);   // aggregation accumulator
    float*    Asrc = (float*)   carve((size_t)NN * 4 * 4);
    float*    Adst = (float*)   carve((size_t)NN * 4 * 4);
    unsigned* Mk   = (unsigned*)carve((size_t)NN * 4 * 4);
    float*    Den  = (float*)   carve((size_t)NN * 4 * 4);
    float*    Ebuf = (float*)   carve((size_t)EDGE * 4 * 4);
    float*    Abuf = (float*)   carve((size_t)EDGE * 4 * 4);

    const int TB = 256;
    auto blks = [](long n, int tb) { return (int)((n + tb - 1) / tb); };

    // layer configs: {Fin, Kp, heads, FoutP(pad64)}
    const int Fin_[5]   = {200, 800, 800, 800, 800};
    const int Kp_[5]    = {224, 800, 800, 800, 800};
    const int heads_[5] = {4, 4, 4, 4, 1};
    const int FoutP_[5] = {832, 832, 832, 832, 256};

    // initial bf16 conversion of X (pad 200 -> 224 along K)
    k_x_to_bf16<<<blks((long)NN * 224, TB), TB, 0, stream>>>(X, Xb, 200, 224);

    float* xFinal  = (float*)d_out;                    // [N,200]
    float* attnOut = (float*)d_out + (size_t)NN * CC;  // [110000]

    for (int l = 0; l < 5; ++l) {
        const int Fin = Fin_[l], Kp = Kp_[l], heads = heads_[l];
        const int Fout = heads * CC, FoutP = FoutP_[l];

        // weights: transpose + bf16, padded
        k_wt_bf16<<<blks((long)FoutP * Kp, TB), TB, 0, stream>>>(W[l], Wt, Fin, Fout, Kp, FoutP);

        // GEMM via WMMA: grid (FoutP/64 col groups, ceil(313/8) row groups)
        dim3 gg(FoutP / 64, (RT32 + 7) / 8);
        k_gemm_wmma<<<gg, TB, 0, stream>>>(Xb, Wt, Hbuf, Kp, FoutP);

        // attention logits per node
        k_alpha<<<blks((long)NN * heads, TB), TB, 0, stream>>>(Hbuf, AS[l], AD[l], Asrc, Adst, heads, FoutP);

        // reset segment-softmax state + output accumulator
        k_fill_u32<<<blks((long)NN * heads, TB), TB, 0, stream>>>(Mk, 0u, NN * heads);
        k_fill_u32<<<blks((long)NN * heads, TB), TB, 0, stream>>>((unsigned*)Den, 0u, NN * heads);
        k_fill_u32<<<blks((long)NN * FoutP, TB), TB, 0, stream>>>((unsigned*)Obuf, 0u, NN * FoutP);

        // segment softmax over incoming edges
        k_edge_max <<<blks((long)EDGE * heads, TB), TB, 0, stream>>>(EI, Asrc, Adst, Ebuf, Mk, heads);
        k_edge_exp <<<blks((long)EDGE * heads, TB), TB, 0, stream>>>(EI, Ebuf, Mk, Den, heads);
        k_edge_norm<<<blks((long)EDGE * heads, TB), TB, 0, stream>>>(EI, Ebuf, Den, Abuf, heads,
                                                                     (l == 4) ? attnOut : nullptr);

        // weighted scatter aggregation
        k_aggregate<<<blks((long)EDGE * heads * (CC / 4), TB), TB, 0, stream>>>(EI, Hbuf, Abuf, Obuf, heads, FoutP);

        // bias + gelu; emit next-layer bf16 activations or final f32 output
        k_bias_gelu<<<blks((long)NN * Fout, TB), TB, 0, stream>>>(
            Obuf, B[l],
            (l < 4) ? Xb : nullptr,
            (l == 4) ? xFinal : nullptr,
            heads, FoutP);
    }
}
